// H2GCN_Net_8323646620245
// MI455X (gfx1250) — matrix operand order repaired and checked by reading, not compiled, not run on hardware
//
#include <hip/hip_runtime.h>
#include <hip/hip_bf16.h>
#include <math.h>

// ---------------------------------------------------------------------------
// H2GCN forward for MI455X (gfx1250, wave32, WMMA).
//
// r_final [100000, 448] fp32 lives in d_ws (179.2 MB -- fits the 192 MB L2):
//   cols   0.. 63 : r0 = relu(x @ w_embed)          (bf16 WMMA, fp32 acc)
//   cols  64..127 : spmm(a1, r0)                    (f32 atomics)
//   cols 128..191 : spmm(a2, r0)
//   cols 192..319 : spmm(a1, r_final[:,64:192])
//   cols 320..447 : spmm(a2, r_final[:,64:192])
// out = log_softmax(r_final @ w_classify)           (bf16 WMMA + shfl reduce)
//
// v2: embed kernel stages the 64 KB B-fragment buffer into LDS once per
// workgroup via global_load_async_to_lds_b128 (ASYNCcnt path); all 8 waves
// then read B from LDS (ds_load_b128) instead of re-reading L2 every K-step.
// ---------------------------------------------------------------------------

typedef __attribute__((ext_vector_type(16))) __bf16 v16bf;
typedef __attribute__((ext_vector_type(8)))  float  v8f;
typedef __attribute__((ext_vector_type(4)))  float  f4;

#define NNODES 100000
#define FDIM   512
#define HDIM   64
#define CDIM   10
#define LDR    448                 // 7*H, row stride of r_final
#define NTILES (NNODES / 16)       // 6250 full 16-row tiles
#define WFRAG_BYTES ((FDIM / 32) * 4 * 32 * 16 * 2)   // 65536

// ---- A-fragment gather: 16x32 bf16 tile slice for one lane ----------------
// Lanes 0-15 hold row M=lane, K = {0..7, 16..23}; lanes 16-31 hold row
// M=lane-16, K = {8..15, 24..31}.  With kb = kk*32 + half*8 the lane needs
// floats [kb..kb+7] and [kb+16..kb+23] of its row.  Non-temporal: these are
// streaming reads (x / r_final) and must not evict the L2-resident state.
__device__ __forceinline__ v16bf load_a_frag_nt(const float* __restrict__ rowp, int kb) {
  float tmp[16] __attribute__((aligned(16)));
  *(f4*)(tmp + 0)  = __builtin_nontemporal_load((const f4*)(rowp + kb));
  *(f4*)(tmp + 4)  = __builtin_nontemporal_load((const f4*)(rowp + kb + 4));
  *(f4*)(tmp + 8)  = __builtin_nontemporal_load((const f4*)(rowp + kb + 16));
  *(f4*)(tmp + 12) = __builtin_nontemporal_load((const f4*)(rowp + kb + 20));
  v16bf a;
#pragma unroll
  for (int i = 0; i < 16; ++i) a[i] = (__bf16)tmp[i];
  return a;
}

__device__ __forceinline__ void wait_asynccnt0() {
#if __has_builtin(__builtin_amdgcn_s_wait_asynccnt)
  __builtin_amdgcn_s_wait_asynccnt(0);
#else
  asm volatile("s_wait_asynccnt 0x0" ::: "memory");
#endif
}

// ---- Pack w_embed (fp32 [512,64]) into WMMA B-fragment layout in scratch --
// B 32x16 bf16: lanes 0-15 = col N=lane, K={0..7,16..23}; lanes 16-31 = N=lane-16,
// K={8..15,24..31}.  frag index = ((kk*4 + ct)*32 + lane)*16, one v16bf per lane.
__global__ void prep_wembed_kernel(const float* __restrict__ w, __bf16* __restrict__ frag) {
  int t = blockIdx.x * blockDim.x + threadIdx.x;        // 16*4*32 = 2048
  if (t >= (FDIM / 32) * 4 * 32) return;
  int lane = t & 31;
  int ct   = (t >> 5) & 3;
  int kk   = t >> 7;
  int col  = ct * 16 + (lane & 15);
  int kb   = kk * 32 + (lane >> 4) * 8;
  __bf16* dst = frag + (size_t)t * 16;                  // t == ((kk*4+ct)*32+lane)
#pragma unroll
  for (int i = 0; i < 16; ++i) dst[i] = (__bf16)w[(size_t)(kb + i + (i >= 8 ? 8 : 0)) * HDIM + col];
}

// ---- Pack w_classify (fp32 [448,10]) into B-fragments, N padded to 16 -----
__global__ void prep_wcls_kernel(const float* __restrict__ w, __bf16* __restrict__ frag) {
  int t = blockIdx.x * blockDim.x + threadIdx.x;        // 14*32 = 448
  if (t >= (LDR / 32) * 32) return;
  int lane = t & 31;
  int kk   = t >> 5;
  int col  = lane & 15;
  int kb   = kk * 32 + (lane >> 4) * 8;
  __bf16* dst = frag + (size_t)t * 16;
#pragma unroll
  for (int i = 0; i < 16; ++i)
    dst[i] = (col < CDIM) ? (__bf16)w[(size_t)(kb + i + (i >= 8 ? 8 : 0)) * CDIM + col] : (__bf16)0.f;
}

// ---- r0 = relu(x @ w_embed): one wave -> 16x64 tile, 16 K-steps, 4 accs ---
__global__ void __launch_bounds__(256) embed_gemm_kernel(
    const float* __restrict__ x, const __bf16* __restrict__ wfrag,
    float* __restrict__ rfinal) {
  extern __shared__ __bf16 sB[];                        // 64 KB dynamic LDS

  // Stage all B fragments into LDS with the async DMA path (no VGPR bounce):
  // 4096 x b128 transfers, 16 per thread.  ASYNCcnt-tracked.
  {
    const char* gsrc = (const char*)wfrag;
    unsigned lds_base = (unsigned)(uintptr_t)(void*)sB; // low 32 bits = LDS offset
#pragma unroll
    for (int i = 0; i < 16; ++i) {
      int byteoff = (i * 256 + (int)threadIdx.x) * 16;
      unsigned lds = lds_base + (unsigned)byteoff;
      const char* g = gsrc + byteoff;
      asm volatile("global_load_async_to_lds_b128 %0, %1, off"
                   :: "v"(lds), "v"(g) : "memory");
    }
    wait_asynccnt0();
  }
  __syncthreads();                                      // all waves see sB

  int wave = blockIdx.x * (blockDim.x >> 5) + (threadIdx.x >> 5);
  if (wave >= NTILES) return;                           // after the only barrier
  int lane = threadIdx.x & 31;
  int half = lane >> 4;
  const float* xrow = x + (size_t)((wave << 4) + (lane & 15)) * FDIM;

  v8f c0 = {}, c1 = {}, c2 = {}, c3 = {};
  for (int kk = 0; kk < FDIM / 32; ++kk) {
    int kb = kk * 32 + half * 8;
    __builtin_prefetch(xrow + kb + 64, 0, 1);           // global_prefetch_b8
    v16bf a = load_a_frag_nt(xrow, kb);
    const v16bf* bp = (const v16bf*)sB + (size_t)kk * 128 + lane;   // ds_load
    c0 = __builtin_amdgcn_wmma_f32_16x16x32_bf16(false, a, false, bp[0],  (short)0, c0, false, false);
    c1 = __builtin_amdgcn_wmma_f32_16x16x32_bf16(false, a, false, bp[32], (short)0, c1, false, false);
    c2 = __builtin_amdgcn_wmma_f32_16x16x32_bf16(false, a, false, bp[64], (short)0, c2, false, false);
    c3 = __builtin_amdgcn_wmma_f32_16x16x32_bf16(false, a, false, bp[96], (short)0, c3, false, false);
  }
  // C/D layout: VGPR j -> row j + 8*half, lanes&15 -> column
#pragma unroll
  for (int j = 0; j < 8; ++j) {
    float* p = rfinal + (size_t)((wave << 4) + half * 8 + j) * LDR + (lane & 15);
    p[0]  = fmaxf(c0[j], 0.f);
    p[16] = fmaxf(c1[j], 0.f);
    p[32] = fmaxf(c2[j], 0.f);
    p[48] = fmaxf(c3[j], 0.f);
  }
}

// ---- SpMM: out[dst,:] += val * in[src,:]; float4 gather + 4 f32 atomics ---
// Random edges, unsorted -> L2 atomic scatter is the right primitive; the
// gathered rows are L2-resident (r_final fits in the 192 MB L2).  atomicAdd
// results are unused -> no-return global_atomic_add_f32 (STOREcnt path).
__global__ void __launch_bounds__(256) spmm_kernel(
    const int* __restrict__ idx, const float* __restrict__ val,
    const float* __restrict__ rin, float* __restrict__ rout,
    int E, int lh /* log2(H/4) */) {
  int t = blockIdx.x * blockDim.x + threadIdx.x;
  int e = t >> lh;
  if (e >= E) return;
  int c = (t & ((1 << lh) - 1)) << 2;
  int dst = idx[e];
  int src = idx[E + e];
  float v = val[e];
  f4 r = *(const f4*)(rin + (size_t)src * LDR + c);
  float* p = rout + (size_t)dst * LDR + c;
  atomicAdd(p + 0, v * r[0]);
  atomicAdd(p + 1, v * r[1]);
  atomicAdd(p + 2, v * r[2]);
  atomicAdd(p + 3, v * r[3]);
}

// ---- logits = r_final @ w_cls (N padded to 16), fused log_softmax --------
__global__ void __launch_bounds__(256) classify_kernel(
    const float* __restrict__ rfinal, const __bf16* __restrict__ wcf,
    float* __restrict__ out) {
  int wave = blockIdx.x * (blockDim.x >> 5) + (threadIdx.x >> 5);
  if (wave >= NTILES) return;
  int lane = threadIdx.x & 31;
  int half = lane >> 4;
  int n = lane & 15;
  const float* arow = rfinal + (size_t)((wave << 4) + n) * LDR;  // A row = lane&15

  v8f acc = {};
  for (int kk = 0; kk < LDR / 32; ++kk) {              // 14 K-steps
    v16bf a = load_a_frag_nt(arow, kk * 32 + half * 8);
    v16bf b = ((const v16bf*)wcf)[(size_t)kk * 32 + lane];  // tiny, L2-hot
    acc = __builtin_amdgcn_wmma_f32_16x16x32_bf16(false, a, false, b, (short)0, acc, false, false);
  }

  // log_softmax over the 10 valid columns; row j+8*half lives in acc[j]
  // across the 16 lanes of this half -> xor-shuffle tree stays in-half.
  bool valid = (n < CDIM);
#pragma unroll
  for (int j = 0; j < 8; ++j) {
    float xv = acc[j];
    float m = valid ? xv : -__builtin_inff();
#pragma unroll
    for (int mask = 8; mask >= 1; mask >>= 1) m = fmaxf(m, __shfl_xor(m, mask, 32));
    float ev = valid ? __expf(xv - m) : 0.f;
    float s = ev;
#pragma unroll
    for (int mask = 8; mask >= 1; mask >>= 1) s += __shfl_xor(s, mask, 32);
    if (valid) {
      int row = (wave << 4) + half * 8 + j;
      out[(size_t)row * CDIM + n] = xv - m - __logf(s);
    }
  }
}

extern "C" void kernel_launch(void* const* d_in, const int* in_sizes, int n_in,
                              void* d_out, int out_size, void* d_ws, size_t ws_size,
                              hipStream_t stream) {
  const float* x       = (const float*)d_in[0];
  const int*   a1_idx  = (const int*)d_in[1];
  const float* a1_val  = (const float*)d_in[2];
  const int*   a2_idx  = (const int*)d_in[3];
  const float* a2_val  = (const float*)d_in[4];
  const float* w_embed = (const float*)d_in[5];
  const float* w_cls   = (const float*)d_in[6];
  const int E1 = in_sizes[2];                 // 1,600,000
  const int E2 = in_sizes[4];                 // 3,200,000

  float* rfinal = (float*)d_ws;
  const size_t rbytes = (size_t)NNODES * LDR * sizeof(float);   // 179.2 MB
  __bf16* wefrag = (__bf16*)((char*)d_ws + rbytes);             // 64 KB
  __bf16* wcfrag = wefrag + (size_t)WFRAG_BYTES / 2;            // 14 KB

  // zero every call: spmm accumulates with atomics (deterministic replay)
  (void)hipMemsetAsync(rfinal, 0, rbytes, stream);

  prep_wembed_kernel<<<8, 256, 0, stream>>>(w_embed, wefrag);
  prep_wcls_kernel<<<2, 256, 0, stream>>>(w_cls, wcfrag);

  embed_gemm_kernel<<<(NTILES + 7) / 8, 256, WFRAG_BYTES, stream>>>(x, wefrag, rfinal);

  // round 1: H = 64, read cols 0..63
  {
    const int lh = 4;  // H/4 = 16 threads per edge
    unsigned b1 = (unsigned)((((long long)E1 << lh) + 255) >> 8);
    unsigned b2 = (unsigned)((((long long)E2 << lh) + 255) >> 8);
    spmm_kernel<<<b1, 256, 0, stream>>>(a1_idx, a1_val, rfinal,      rfinal + 64,  E1, lh);
    spmm_kernel<<<b2, 256, 0, stream>>>(a2_idx, a2_val, rfinal,      rfinal + 128, E2, lh);
  }
  // round 2: H = 128, read cols 64..191
  {
    const int lh = 5;  // H/4 = 32 threads per edge
    unsigned b1 = (unsigned)((((long long)E1 << lh) + 255) >> 8);
    unsigned b2 = (unsigned)((((long long)E2 << lh) + 255) >> 8);
    spmm_kernel<<<b1, 256, 0, stream>>>(a1_idx, a1_val, rfinal + 64, rfinal + 192, E1, lh);
    spmm_kernel<<<b2, 256, 0, stream>>>(a2_idx, a2_val, rfinal + 64, rfinal + 320, E2, lh);
  }

  classify_kernel<<<(NTILES + 7) / 8, 256, 0, stream>>>(rfinal, wcfrag, (float*)d_out);
}
